// SelfAttentionPooling_33414845563021
// MI455X (gfx1250) — matrix operand order repaired and checked by reading, not compiled, not run on hardware
//
#include <hip/hip_runtime.h>
#include <hip/hip_bf16.h>
#include <math.h>

#define Bsz 16
#define Nsz 2048
#define Dsz 128
#define Osz 128

typedef __attribute__((ext_vector_type(16))) __bf16        v16bf;
typedef __attribute__((ext_vector_type(8)))  float         v8f;
typedef __attribute__((ext_vector_type(4)))  unsigned int  v4u;
typedef __attribute__((ext_vector_type(4)))  float         v4f;

union FragB {
  v16bf v;
  unsigned short u[16];
  v4u q[2];
};

__device__ __forceinline__ unsigned short f32_to_bf16(float f) {
  union { float f; unsigned u; } a; a.f = f;
  unsigned r = a.u + 0x7FFFu + ((a.u >> 16) & 1u);   // round-to-nearest-even
  return (unsigned short)(r >> 16);
}

// ---------------------------------------------------------------------------
// Kernel 1: weight norm.  W[i,:] = g[i] * v[i,:] / ||v[i,:]||
// mats 0..2 (q,k,v) -> bf16 row-major; mat 3 (o) -> f32 row-major.
// ---------------------------------------------------------------------------
__global__ __launch_bounds__(128)
void wn_kernel(const float* __restrict__ vq, const float* __restrict__ gq,
               const float* __restrict__ vk, const float* __restrict__ gk,
               const float* __restrict__ vv, const float* __restrict__ gv,
               const float* __restrict__ vo, const float* __restrict__ go,
               unsigned short* __restrict__ Wq, unsigned short* __restrict__ Wk,
               unsigned short* __restrict__ Wv, float* __restrict__ Wo) {
  __shared__ float s[128];
  int mat = blockIdx.x >> 7;
  int row = blockIdx.x & 127;
  int t = threadIdx.x;
  const float* v; const float* g;
  if (mat == 0)      { v = vq; g = gq; }
  else if (mat == 1) { v = vk; g = gk; }
  else if (mat == 2) { v = vv; g = gv; }
  else               { v = vo; g = go; }
  float x = v[row * Dsz + t];
  s[t] = x * x;
  __syncthreads();
  for (int off = 64; off > 0; off >>= 1) {
    if (t < off) s[t] += s[t + off];
    __syncthreads();
  }
  float scale = g[row] * rsqrtf(s[0]);
  float w = x * scale;
  if (mat == 0)      Wq[row * Dsz + t] = f32_to_bf16(w);
  else if (mat == 1) Wk[row * Dsz + t] = f32_to_bf16(w);
  else if (mat == 2) Wv[row * Dsz + t] = f32_to_bf16(w);
  else               Wo[row * Dsz + t] = w;
}

__global__ void zero_kernel(float* __restrict__ p, int n) {
  int i = blockIdx.x * blockDim.x + threadIdx.x;
  if (i < n) p[i] = 0.f;
}

// ---------------------------------------------------------------------------
// Kernel 2: Q/K/V projections via v_wmma_f32_16x16x32_bf16.
// blockIdx.y selects the matrix (uniform -> scalar branches only).
// Loop nest is kc-outer / jo-inner: 8 independent accumulator chains, so
// consecutive WMMAs never share a D register (no RAW hazard NOPs).
// ---------------------------------------------------------------------------
__global__ __launch_bounds__(256)
void proj_kernel(const float* __restrict__ x,
                 const unsigned short* __restrict__ Wq,
                 const unsigned short* __restrict__ Wk,
                 const unsigned short* __restrict__ Wv,
                 const float* __restrict__ bq,
                 const float* __restrict__ bk,
                 const float* __restrict__ bv,
                 unsigned short* __restrict__ Qo,
                 unsigned short* __restrict__ Ko,
                 float* __restrict__ Vo) {
  int wid  = threadIdx.x >> 5;
  int lane = threadIdx.x & 31;
  int mat  = blockIdx.y;                      // uniform: 0=Q 1=K 2=V
  int unit = blockIdx.x * 8 + wid;            // 0 .. 2047
  int tile = unit & 127;                      // 16-row tile within batch
  int b    = unit >> 7;
  int laneHi = (lane >= 16) ? 1 : 0;
  int mrow = lane & 15;
  const float* xr = x + ((size_t)b * Nsz + (size_t)tile * 16 + mrow) * Dsz;

  // A fragments (16x32 bf16): per lane K = kb..kb+7 and kb+16..kb+23, kb = 8*laneHi
  FragB a[4];
  int kb = laneHi ? 8 : 0;
#pragma unroll
  for (int kc = 0; kc < 4; ++kc) {
    const v4f* p0 = (const v4f*)(xr + kc * 32 + kb);
    const v4f* p1 = (const v4f*)(xr + kc * 32 + kb + 16);
    v4f f0 = p0[0], f1 = p0[1];
    v4f f2 = p1[0], f3 = p1[1];
#pragma unroll
    for (int e = 0; e < 4; ++e) {
      a[kc].u[e]      = f32_to_bf16(f0[e]);
      a[kc].u[4 + e]  = f32_to_bf16(f1[e]);
      a[kc].u[8 + e]  = f32_to_bf16(f2[e]);
      a[kc].u[12 + e] = f32_to_bf16(f3[e]);
    }
  }

  const unsigned short* W = (mat == 0) ? Wq : ((mat == 1) ? Wk : Wv);
  const float* bias       = (mat == 0) ? bq : ((mat == 1) ? bk : bv);
  int nsel = lane & 15;
  int dinB = laneHi ? 8 : 0;                  // B 32x16: per lane K = dinB..dinB+7 and dinB+16..dinB+23

  v8f acc[8];
#pragma unroll
  for (int jo = 0; jo < 8; ++jo) acc[jo] = (v8f){0.f,0.f,0.f,0.f,0.f,0.f,0.f,0.f};

#pragma unroll
  for (int kc = 0; kc < 4; ++kc) {
#pragma unroll
    for (int jo = 0; jo < 8; ++jo) {          // independent D regs back-to-back
      FragB bf;
      const v4u* wp = (const v4u*)(W + (size_t)(jo * 16 + nsel) * Dsz + kc * 32 + dinB);
      bf.q[0] = wp[0];
      bf.q[1] = wp[2];
      acc[jo] = __builtin_amdgcn_wmma_f32_16x16x32_bf16(false, a[kc].v, false, bf.v,
                                                        (short)0, acc[jo], false, false);
    }
  }

  size_t rowBase = ((size_t)b * Nsz + (size_t)tile * 16 + (laneHi ? 8 : 0)) * Dsz + nsel;
  if (mat == 0) {                             // uniform scalar branch
#pragma unroll
    for (int jo = 0; jo < 8; ++jo) {
      float bsv = bias[jo * 16 + nsel];
#pragma unroll
      for (int r = 0; r < 8; ++r)
        Qo[rowBase + jo * 16 + (size_t)r * Dsz] = f32_to_bf16(acc[jo][r] + bsv);
    }
  } else if (mat == 1) {
#pragma unroll
    for (int jo = 0; jo < 8; ++jo) {
      float bsv = bias[jo * 16 + nsel];
#pragma unroll
      for (int r = 0; r < 8; ++r)
        Ko[rowBase + jo * 16 + (size_t)r * Dsz] = f32_to_bf16(acc[jo][r] + bsv);
    }
  } else {
#pragma unroll
    for (int jo = 0; jo < 8; ++jo) {
      float bsv = bias[jo * 16 + nsel];
#pragma unroll
      for (int r = 0; r < 8; ++r)
        Vo[rowBase + jo * 16 + (size_t)r * Dsz] = acc[jo][r] + bsv;
    }
  }
}

// ---------------------------------------------------------------------------
// Kernel 3: scores + softmax + column-sum of attention weights.
// One WG (256 thr / 8 waves) per (batch, 16-query-row tile).
// Full 16x2048 f32 score tile in dynamic LDS (128KB of the 320KB WGP LDS).
// Two col-tiles in flight per iteration -> alternating WMMA chains hide the
// WMMA->WMMA RAW hazard slots.  c[b,m] += sum_n exp(S[n,m]-M_n)/L_n.
// ---------------------------------------------------------------------------
__global__ __launch_bounds__(256)
void attn_colsum_kernel(const unsigned short* __restrict__ Q,
                        const unsigned short* __restrict__ K,
                        const int* __restrict__ mask,
                        float* __restrict__ csum) {
  extern __shared__ float S[];                // 16*Nsz scores
  float* rowM = S + 16 * Nsz;                 // 16
  float* rowL = rowM + 16;                    // 16
  float* red  = rowL + 16;                    // 256 scratch

  int b    = blockIdx.x >> 7;                 // 128 tiles per batch
  int tile = blockIdx.x & 127;
  int lane = threadIdx.x & 31;
  int wid  = threadIdx.x >> 5;
  int laneHi = (lane >= 16) ? 1 : 0;
  int t = threadIdx.x;

  // Preload the 4 Q A-fragments for this row tile (bf16, two 16B loads each)
  FragB a[4];
  {
    int mrow = lane & 15;
    const unsigned short* qr = Q + ((size_t)b * Nsz + (size_t)tile * 16 + mrow) * Dsz;
    int kb = laneHi ? 8 : 0;
#pragma unroll
    for (int kc = 0; kc < 4; ++kc) {
      const v4u* p0 = (const v4u*)(qr + kc * 32 + kb);
      const v4u* p1 = (const v4u*)(qr + kc * 32 + kb + 16);
      a[kc].q[0] = p0[0];
      a[kc].q[1] = p1[0];
    }
  }

  // S = Q K^T : 8 waves sweep 128 key col-tiles, 2 tiles per iteration
  int db   = laneHi ? 8 : 0;
  int nsel = lane & 15;
  for (int ct = wid; ct < Nsz / 16; ct += 16) {
    const unsigned short* kr0 = K + ((size_t)b * Nsz + ct * 16 + nsel) * Dsz;
    const unsigned short* kr1 = kr0 + (size_t)8 * 16 * Dsz;      // tile ct+8
    if (ct + 16 < Nsz / 16) {                 // wave-uniform guard
      // pull next pair of key tiles toward the WGP (global_prefetch_b8)
      __builtin_prefetch(kr0 + (size_t)16 * 16 * Dsz, 0, 3);
      __builtin_prefetch(kr1 + (size_t)16 * 16 * Dsz, 0, 3);
    }
    v8f acc0 = {0.f,0.f,0.f,0.f,0.f,0.f,0.f,0.f};
    v8f acc1 = {0.f,0.f,0.f,0.f,0.f,0.f,0.f,0.f};
#pragma unroll
    for (int kc = 0; kc < 4; ++kc) {
      FragB bf0, bf1;
      const v4u* p0 = (const v4u*)(kr0 + kc * 32 + db);
      const v4u* p1 = (const v4u*)(kr1 + kc * 32 + db);
      bf0.q[0] = p0[0]; bf0.q[1] = p0[2];
      bf1.q[0] = p1[0]; bf1.q[1] = p1[2];
      acc0 = __builtin_amdgcn_wmma_f32_16x16x32_bf16(false, a[kc].v, false, bf0.v,
                                                     (short)0, acc0, false, false);
      acc1 = __builtin_amdgcn_wmma_f32_16x16x32_bf16(false, a[kc].v, false, bf1.v,
                                                     (short)0, acc1, false, false);
    }
#pragma unroll
    for (int r = 0; r < 8; ++r) {
      int rl = r + (laneHi ? 8 : 0);
      S[rl * Nsz + ct * 16 + nsel]         = acc0[r];
      S[rl * Nsz + (ct + 8) * 16 + nsel]   = acc1[r];
    }
  }
  __syncthreads();

  // Apply pair mask in place: scores = pair ? S : -inf
  const int* mb = mask + b * Nsz;
  int* rowMaskI = (int*)red;
  if (t < 16) rowMaskI[t] = mb[tile * 16 + t];
  __syncthreads();
  for (int idx = t; idx < 16 * Nsz; idx += 256) {
    int row = idx >> 11;                      // Nsz == 2048
    int col = idx & (Nsz - 1);
    if (!(rowMaskI[row] & mb[col])) S[idx] = -__builtin_inff();
  }
  __syncthreads();

  // Row max (16 threads per row)
  int rr  = t >> 4;
  int sub = t & 15;
  float mx = -__builtin_inff();
  for (int c = sub; c < Nsz; c += 16) mx = fmaxf(mx, S[rr * Nsz + c]);
  red[t] = mx;
  __syncthreads();
  if (sub == 0) {
    float m2 = red[t];
    for (int i = 1; i < 16; ++i) m2 = fmaxf(m2, red[t + i]);
    rowM[rr] = m2;
  }
  __syncthreads();

  // Row sum of exp
  float M = rowM[rr];
  float sm = 0.f;
  for (int c = sub; c < Nsz; c += 16) sm += __expf(S[rr * Nsz + c] - M);
  red[t] = sm;
  __syncthreads();
  if (sub == 0) {
    float s2 = 0.f;
    for (int i = 0; i < 16; ++i) s2 += red[t + i];
    rowL[rr] = 1.0f / s2;                     // store reciprocal
  }
  __syncthreads();

  // Column sums of normalized weights -> global accumulate
  for (int c = t; c < Nsz; c += 256) {
    float accc = 0.f;
#pragma unroll 4
    for (int r = 0; r < 16; ++r)
      accc += __expf(S[r * Nsz + c] - rowM[r]) * rowL[r];
    atomicAdd(&csum[b * Nsz + c], accc);
  }
}

// ---------------------------------------------------------------------------
// Kernel 4: pooled = c @ V ; out = pooled @ Wo^T + b_o.   One block per batch.
// ---------------------------------------------------------------------------
__global__ __launch_bounds__(128)
void final_kernel(const float* __restrict__ csum,
                  const float* __restrict__ V,
                  const float* __restrict__ Wo,
                  const float* __restrict__ bo,
                  float* __restrict__ out) {
  __shared__ float pooled[Dsz];
  int b = blockIdx.x;
  int t = threadIdx.x;
  const float* cb = csum + b * Nsz;
  const float* vb = V + (size_t)b * Nsz * Dsz;
  float acc = 0.f;
  for (int m = 0; m < Nsz; ++m) acc += cb[m] * vb[(size_t)m * Dsz + t];
  pooled[t] = acc;
  __syncthreads();
  float o = bo[t];
  const float* wr = Wo + (size_t)t * Dsz;
#pragma unroll 8
  for (int d = 0; d < Dsz; ++d) o += pooled[d] * wr[d];
  out[b * Osz + t] = o;
}

// ---------------------------------------------------------------------------
extern "C" void kernel_launch(void* const* d_in, const int* in_sizes, int n_in,
                              void* d_out, int out_size, void* d_ws, size_t ws_size,
                              hipStream_t stream) {
  const float* x   = (const float*)d_in[0];
  const int* mask  = (const int*)d_in[1];
  const float* v_q = (const float*)d_in[2];
  const float* g_q = (const float*)d_in[3];
  const float* b_q = (const float*)d_in[4];
  const float* v_k = (const float*)d_in[5];
  const float* g_k = (const float*)d_in[6];
  const float* b_k = (const float*)d_in[7];
  const float* v_v = (const float*)d_in[8];
  const float* g_v = (const float*)d_in[9];
  const float* b_v = (const float*)d_in[10];
  const float* v_o = (const float*)d_in[11];
  const float* g_o = (const float*)d_in[12];
  const float* b_o = (const float*)d_in[13];

  char* ws = (char*)d_ws;
  const size_t offWq = 0;
  const size_t offWk = 32768;                         // 128*128*2
  const size_t offWv = 65536;
  const size_t offWo = 98304;                         // 128*128*4
  const size_t offQ  = 163840;
  const size_t offK  = offQ + (size_t)Bsz * Nsz * Dsz * 2;   // +8MB
  const size_t offV  = offK + (size_t)Bsz * Nsz * Dsz * 2;   // +8MB
  const size_t offC  = offV + (size_t)Bsz * Nsz * Dsz * 4;   // +16MB

  unsigned short* Wq = (unsigned short*)(ws + offWq);
  unsigned short* Wk = (unsigned short*)(ws + offWk);
  unsigned short* Wv = (unsigned short*)(ws + offWv);
  float* Wo          = (float*)(ws + offWo);
  unsigned short* Q  = (unsigned short*)(ws + offQ);
  unsigned short* Kp = (unsigned short*)(ws + offK);
  float* V           = (float*)(ws + offV);
  float* csum        = (float*)(ws + offC);
  float* out         = (float*)d_out;

  wn_kernel<<<512, 128, 0, stream>>>(v_q, g_q, v_k, g_k, v_v, g_v, v_o, g_o,
                                     Wq, Wk, Wv, Wo);
  zero_kernel<<<(Bsz * Nsz) / 256, 256, 0, stream>>>(csum, Bsz * Nsz);
  proj_kernel<<<dim3(256, 3, 1), 256, 0, stream>>>(x, Wq, Wk, Wv, b_q, b_k, b_v,
                                                   Q, Kp, V);
  size_t smem = (size_t)16 * Nsz * 4 + (16 + 16 + 256) * 4;  // 132224 B
  attn_colsum_kernel<<<Bsz * (Nsz / 16), 256, smem, stream>>>(Q, Kp, mask, csum);
  final_kernel<<<Bsz, 128, 0, stream>>>(csum, V, Wo, b_o, out);
}
